// GraphContrastiveModel_90829968376151
// MI455X (gfx1250) — matrix operand order repaired and checked by reading, not compiled, not run on hardware
//
#include <hip/hip_runtime.h>
#include <hip/hip_bf16.h>

// ---------------------------------------------------------------------------
// Problem constants (from reference)
// ---------------------------------------------------------------------------
#define NNODES   81920
#define NEDGES   180224
#define NGRAPHS  2048
#define EMB      300
#define EMBP     320      // padded K / channel dim (mult of 32)
#define HID      600
#define HIDP     608      // padded hidden dim (19*32)
#define NLAYERS  5
#define FEATROWS 4096     // 2 * NGRAPHS
#define INV_TEMP 25.0f    // 1 / 0.04
#define BN_EPS   1e-5f
#define KT_MAX   19       // max Kp/32 (HIDP/32)

typedef __attribute__((ext_vector_type(16))) __bf16 v16bf;
typedef __attribute__((ext_vector_type(8)))  float  v8f;

union FragAB { uint4 q[2]; v16bf v; };

__device__ __forceinline__ unsigned short f2bf(float x) {
    unsigned u = __float_as_uint(x);
    u += 0x7fffu + ((u >> 16) & 1u);           // round to nearest even
    return (unsigned short)(u >> 16);
}

// ---------------------------------------------------------------------------
// WMMA bf16 GEMM:  C[M,Np] = A[M,Kp] * B[Kp,Np] (+bias) (relu?)
// A: row-major bf16 (ushort), Kp multiple of 32.
// Bpk: packed B fragments: [((nt*KT + kt)*32 + lane)*16 + i]
//      lane<16 : column nt*16+lane,    K = kt*32 + (i<8 ? i : i+8)  (i = 0..15)
//      lane>=16: column nt*16+lane-16, K = kt*32 + 8 + (i<8 ? i : i+8)
// Block = 128 threads (4 waves); block tile = 128(M) x 32(N).
// The 2*KT B-fragment panel (contiguous in Bpk) is staged into LDS once per
// block with global_load_async_to_lds_b128 (ASYNCcnt), then each wave computes
// a 32x32 tile: 2 A-frags x 2 B-frags -> 4 WMMAs per k-step.
// Requires M % 128 == 0 (no partial blocks; cooperative barrier inside).
// ---------------------------------------------------------------------------
__global__ void __launch_bounds__(128)
gemm_bf16_wmma(const unsigned short* __restrict__ A,
               const unsigned short* __restrict__ Bpk,
               const float* __restrict__ bias,
               float* __restrict__ Cf,
               unsigned short* __restrict__ Cb,
               int M, int Kp, int Np, int ldC, int relu) {
    __shared__ unsigned short Bsh[2 * KT_MAX * 512];   // 38,912 B max

    const int lane = threadIdx.x & 31;
    const int wave = threadIdx.x >> 5;
    const int KT   = Kp >> 5;
    const int m0   = blockIdx.x * 128 + wave * 32;     // wave: rows [m0, m0+32)
    const int n0   = blockIdx.y * 32;

    // ---- cooperative async stage of the B panel (2 n-tiles, all kt) ----
    {
        const unsigned short* bpanel = Bpk + (size_t)(n0 >> 4) * KT * 512;
        const unsigned panelHalfs = (unsigned)(2 * KT * 512);
        unsigned ldsBase = (unsigned)(size_t)&Bsh[0];  // low 32 bits = LDS offset
        for (unsigned i = threadIdx.x * 8u; i < panelHalfs; i += 128u * 8u) {
            unsigned long long ga = (unsigned long long)(size_t)(bpanel + i);
            unsigned ld = ldsBase + i * 2u;
            asm volatile("global_load_async_to_lds_b128 %0, %1, off"
                         :: "v"(ld), "v"(ga) : "memory");
        }
        asm volatile("s_wait_asynccnt 0x0" ::: "memory");
    }
    __syncthreads();

    const int lrow  = lane & 15;
    const int hi    = lane >> 4;                // 0 or 1
    const int khalf = hi * 8;

    const unsigned short* arow0 = A + (size_t)(m0 + lrow) * Kp + khalf;
    const unsigned short* arow1 = arow0 + (size_t)16 * Kp;
    const unsigned short* bsh0  = &Bsh[lane * 16];             // n-tile 0
    const unsigned short* bsh1  = &Bsh[(size_t)KT * 512 + lane * 16];

    v8f acc00 = {}, acc01 = {}, acc10 = {}, acc11 = {};
    for (int kt = 0; kt < KT; ++kt) {
        FragAB a0, a1, b0, b1;
        const unsigned short* ap0 = arow0 + kt * 32;
        const unsigned short* ap1 = arow1 + kt * 32;
        a0.q[0] = *(const uint4*)(ap0);
        a0.q[1] = *(const uint4*)(ap0 + 16);
        a1.q[0] = *(const uint4*)(ap1);
        a1.q[1] = *(const uint4*)(ap1 + 16);
        b0.q[0] = *(const uint4*)(bsh0 + (size_t)kt * 512);
        b0.q[1] = *(const uint4*)(bsh0 + (size_t)kt * 512 + 8);
        b1.q[0] = *(const uint4*)(bsh1 + (size_t)kt * 512);
        b1.q[1] = *(const uint4*)(bsh1 + (size_t)kt * 512 + 8);
        __builtin_prefetch(ap0 + 64, 0, 1);     // global_prefetch_b8
        __builtin_prefetch(ap1 + 64, 0, 1);
        acc00 = __builtin_amdgcn_wmma_f32_16x16x32_bf16(false, a0.v, false, b0.v,
                                                        (short)0, acc00, false, false);
        acc01 = __builtin_amdgcn_wmma_f32_16x16x32_bf16(false, a0.v, false, b1.v,
                                                        (short)0, acc01, false, false);
        acc10 = __builtin_amdgcn_wmma_f32_16x16x32_bf16(false, a1.v, false, b0.v,
                                                        (short)0, acc10, false, false);
        acc11 = __builtin_amdgcn_wmma_f32_16x16x32_bf16(false, a1.v, false, b1.v,
                                                        (short)0, acc11, false, false);
    }

    const int ncol0 = n0 + lrow;
    const int ncol1 = n0 + 16 + lrow;
    float bias0 = bias ? bias[ncol0] : 0.f;
    float bias1 = bias ? bias[ncol1] : 0.f;
    for (int r = 0; r < 8; ++r) {
        int mA = m0 + r + hi * 8;               // tile row block 0
        int mB = mA + 16;                       // tile row block 1
        float v00 = acc00[r] + bias0, v01 = acc01[r] + bias1;
        float v10 = acc10[r] + bias0, v11 = acc11[r] + bias1;
        if (relu) {
            v00 = fmaxf(v00, 0.f); v01 = fmaxf(v01, 0.f);
            v10 = fmaxf(v10, 0.f); v11 = fmaxf(v11, 0.f);
        }
        if (mA < M) {
            size_t o0 = (size_t)mA * ldC + ncol0, o1 = (size_t)mA * ldC + ncol1;
            if (Cf) { Cf[o0] = v00; Cf[o1] = v01; }
            if (Cb) { Cb[o0] = f2bf(v00); Cb[o1] = f2bf(v01); }
        }
        if (mB < M) {
            size_t o0 = (size_t)mB * ldC + ncol0, o1 = (size_t)mB * ldC + ncol1;
            if (Cf) { Cf[o0] = v10; Cf[o1] = v11; }
            if (Cb) { Cb[o0] = f2bf(v10); Cb[o1] = f2bf(v11); }
        }
    }
}

// ---------------------------------------------------------------------------
// Pack a (strided) f32 matrix element (k,n) = src[k*sk + n*sn] into the WMMA
// B-fragment layout, zero-padded to [Kp, Np].
// ---------------------------------------------------------------------------
__global__ void pack_b(const float* __restrict__ src, int Kr, int Nr,
                       long sk, long sn, int Kp, int Np,
                       unsigned short* __restrict__ out) {
    size_t idx = (size_t)blockIdx.x * blockDim.x + threadIdx.x;
    size_t total = (size_t)Kp * Np;
    if (idx >= total) return;
    int i    = idx & 15;
    int lane = (int)((idx >> 4) & 31);
    size_t t = idx >> 9;                 // nt*KT + kt
    int KT = Kp >> 5;
    int kt = (int)(t % KT);
    int nt = (int)(t / KT);
    int n = nt * 16 + (lane & 15);
    int k = kt * 32 + ((lane >> 4) ? 8 : 0) + (i < 8 ? i : i + 8);
    unsigned short v = 0;
    if (k < Kr && n < Nr) v = f2bf(src[(size_t)k * sk + (size_t)n * sn]);
    out[idx] = v;
}

__global__ void pad_bias(const float* __restrict__ b, int Nr, int Np,
                         float* __restrict__ out) {
    int i = blockIdx.x * blockDim.x + threadIdx.x;
    if (i < Np) out[i] = (i < Nr) ? b[i] : 0.f;
}

// ---------------------------------------------------------------------------
// Node embedding init: h[n, c] = atom_emb1[x[n,0], c] + atom_emb2[x[n,1], c]
// ---------------------------------------------------------------------------
__global__ void init_h(const int* __restrict__ x, const float* __restrict__ a1,
                       const float* __restrict__ a2, float* __restrict__ h) {
    size_t idx = (size_t)blockIdx.x * blockDim.x + threadIdx.x;
    if (idx >= (size_t)NNODES * EMBP) return;
    int n = (int)(idx / EMBP), c = (int)(idx % EMBP);
    float v = 0.f;
    if (c < EMB) v = a1[(size_t)x[n * 2] * EMB + c] + a2[(size_t)x[n * 2 + 1] * EMB + c];
    h[idx] = v;
}

// agg = h + selfloop_edge_embedding (broadcast per channel)
__global__ void agg_init(const float* __restrict__ h, const float* __restrict__ se1,
                         const float* __restrict__ se2, float* __restrict__ agg) {
    size_t idx = (size_t)blockIdx.x * blockDim.x + threadIdx.x;
    if (idx >= (size_t)NNODES * EMBP) return;
    int c = (int)(idx % EMBP);
    float v = h[idx];
    if (c < EMB) v += se1[c] + se2[c];
    agg[idx] = v;
}

// agg[dst] += h[src] + edge_emb1[ea0] + edge_emb2[ea1]
__global__ void scatter_edges(const int* __restrict__ ei, const int* __restrict__ ea,
                              const float* __restrict__ h,
                              const float* __restrict__ ee1, const float* __restrict__ ee2,
                              float* __restrict__ agg) {
    size_t idx = (size_t)blockIdx.x * blockDim.x + threadIdx.x;
    if (idx >= (size_t)NEDGES * EMB) return;
    int e = (int)(idx / EMB), c = (int)(idx % EMB);
    int src = ei[e], dst = ei[NEDGES + e];
    int a0 = ea[e * 2], a1 = ea[e * 2 + 1];
    float v = h[(size_t)src * EMBP + c] + ee1[(size_t)a0 * EMB + c] + ee2[(size_t)a1 * EMB + c];
    atomicAdd(&agg[(size_t)dst * EMBP + c], v);
}

__global__ void cvt_bf16(const float* __restrict__ in, unsigned short* __restrict__ out,
                         size_t n) {
    size_t idx = (size_t)blockIdx.x * blockDim.x + threadIdx.x;
    if (idx < n) out[idx] = f2bf(in[idx]);
}

// One block per channel; biased mean/var over NNODES rows.
__global__ void bn_stats(const float* __restrict__ x, float* __restrict__ stats) {
    int c = blockIdx.x, t = threadIdx.x;
    __shared__ float s1[256], s2[256];
    float a = 0.f, b = 0.f;
    for (int r = t; r < NNODES; r += 256) {
        float v = x[(size_t)r * EMBP + c];
        a += v; b += v * v;
    }
    s1[t] = a; s2[t] = b; __syncthreads();
    for (int s = 128; s > 0; s >>= 1) {
        if (t < s) { s1[t] += s1[t + s]; s2[t] += s2[t + s]; }
        __syncthreads();
    }
    if (t == 0) {
        float mu = s1[0] / (float)NNODES;
        stats[c] = mu;
        stats[EMBP + c] = s2[0] / (float)NNODES - mu * mu;
    }
}

__global__ void bn_apply(const float* __restrict__ x, const float* __restrict__ stats,
                         const float* __restrict__ g, const float* __restrict__ b,
                         float* __restrict__ h, int relu) {
    size_t idx = (size_t)blockIdx.x * blockDim.x + threadIdx.x;
    if (idx >= (size_t)NNODES * EMBP) return;
    int c = (int)(idx % EMBP);
    float v = 0.f;
    if (c < EMB) {
        v = (x[idx] - stats[c]) * rsqrtf(stats[EMBP + c] + BN_EPS) * g[c] + b[c];
        if (relu) v = fmaxf(v, 0.f);
    }
    h[idx] = v;
}

__global__ void pool_zero(float* __restrict__ g, float* __restrict__ cnt) {
    size_t idx = (size_t)blockIdx.x * blockDim.x + threadIdx.x;
    if (idx < (size_t)NGRAPHS * EMBP) g[idx] = 0.f;
    if (idx < NGRAPHS) cnt[idx] = 0.f;
}

__global__ void pool_scatter(const float* __restrict__ h, const int* __restrict__ batch,
                             float* __restrict__ g, float* __restrict__ cnt) {
    size_t idx = (size_t)blockIdx.x * blockDim.x + threadIdx.x;
    if (idx >= (size_t)NNODES * EMBP) return;
    int n = (int)(idx / EMBP), c = (int)(idx % EMBP);
    int bb = batch[n];
    if (c < EMB) atomicAdd(&g[(size_t)bb * EMBP + c], h[idx]);
    if (c == 0) atomicAdd(&cnt[bb], 1.f);
}

__global__ void pool_div(const float* __restrict__ g, const float* __restrict__ cnt,
                         unsigned short* __restrict__ gB) {
    size_t idx = (size_t)blockIdx.x * blockDim.x + threadIdx.x;
    if (idx >= (size_t)NGRAPHS * EMBP) return;
    int b = (int)(idx / EMBP);
    gB[idx] = f2bf(g[idx] / fmaxf(cnt[b], 1.f));
}

// L2-normalize rows of projF (over first EMB cols) into feat (f32) + featB (bf16)
__global__ void norm_rows(const float* __restrict__ in, float* __restrict__ feat,
                          unsigned short* __restrict__ featB) {
    int r = blockIdx.x, t = threadIdx.x;
    __shared__ float ss[256];
    float a = 0.f;
    for (int c = t; c < EMB; c += 256) { float v = in[(size_t)r * EMBP + c]; a += v * v; }
    ss[t] = a; __syncthreads();
    for (int s = 128; s > 0; s >>= 1) {
        if (t < s) ss[t] += ss[t + s];
        __syncthreads();
    }
    float inv = 1.f / fmaxf(sqrtf(ss[0]), 1e-12f);
    for (int c = t; c < EMBP; c += 256) {
        float v = (c < EMB) ? in[(size_t)r * EMBP + c] * inv : 0.f;
        feat[(size_t)r * EMBP + c] = v;
        featB[(size_t)r * EMBP + c] = f2bf(v);
    }
}

// logits[r,0] = sim[r,pos]/T ; logits[r,1+j] = sim[r,col_j]/T (skip r and pos)
__global__ void make_logits(const float* __restrict__ sim, float* __restrict__ out) {
    size_t idx = (size_t)blockIdx.x * blockDim.x + threadIdx.x;
    size_t total = (size_t)FEATROWS * (FEATROWS - 1);
    if (idx >= total) return;
    int r = (int)(idx / (FEATROWS - 1));
    int j = (int)(idx % (FEATROWS - 1));
    int pos = (r + NGRAPHS) & (FEATROWS - 1);
    int col;
    if (j == 0) col = pos;
    else {
        int c = j - 1;
        int lo = r < pos ? r : pos, hi = r < pos ? pos : r;
        if (c >= lo) ++c;
        if (c >= hi) ++c;
        col = c;
    }
    out[idx] = sim[(size_t)r * FEATROWS + col] * INV_TEMP;
}

__global__ void make_labels(float* __restrict__ out) {
    int i = blockIdx.x * blockDim.x + threadIdx.x;
    if (i < FEATROWS) out[i] = 0.f;
}

// ---------------------------------------------------------------------------
// Host-side orchestration
// ---------------------------------------------------------------------------
static inline size_t bump(size_t& off, size_t bytes) {
    size_t o = off;
    off = (off + bytes + 255) & ~(size_t)255;
    return o;
}

extern "C" void kernel_launch(void* const* d_in, const int* in_sizes, int n_in,
                              void* d_out, int out_size, void* d_ws, size_t ws_size,
                              hipStream_t stream) {
    (void)in_sizes; (void)n_in; (void)out_size; (void)ws_size;
    const int* x[2]     = { (const int*)d_in[0], (const int*)d_in[4] };
    const int* ei[2]    = { (const int*)d_in[1], (const int*)d_in[5] };
    const int* ea[2]    = { (const int*)d_in[2], (const int*)d_in[6] };
    const int* batch[2] = { (const int*)d_in[3], (const int*)d_in[7] };
    const float* atom1 = (const float*)d_in[8];
    const float* atom2 = (const float*)d_in[9];
    const float* ee1   = (const float*)d_in[10];   // [5,6,300]
    const float* ee2   = (const float*)d_in[11];   // [5,3,300]
    const float* W1    = (const float*)d_in[12];   // [5,300,600]
    const float* b1    = (const float*)d_in[13];   // [5,600]
    const float* W2    = (const float*)d_in[14];   // [5,600,300]
    const float* b2    = (const float*)d_in[15];   // [5,300]
    const float* bng   = (const float*)d_in[16];
    const float* bnb   = (const float*)d_in[17];
    const float* pW1   = (const float*)d_in[18];
    const float* pb1   = (const float*)d_in[19];
    const float* pW2   = (const float*)d_in[20];
    const float* pb2   = (const float*)d_in[21];
    float* out = (float*)d_out;
    char* ws = (char*)d_ws;

    // workspace layout
    size_t off = 0;
    size_t o_h     = bump(off, (size_t)NNODES * EMBP * 4);   // also aliases sim
    size_t o_agg   = bump(off, (size_t)NNODES * EMBP * 4);
    size_t o_aggB  = bump(off, (size_t)NNODES * EMBP * 2);
    size_t o_midB  = bump(off, (size_t)NNODES * HIDP * 2);
    size_t o_stats = bump(off, (size_t)2 * EMBP * 4);
    size_t o_g     = bump(off, (size_t)NGRAPHS * EMBP * 4);
    size_t o_cnt   = bump(off, (size_t)NGRAPHS * 4);
    size_t o_gB    = bump(off, (size_t)NGRAPHS * EMBP * 2);
    size_t o_p1B   = bump(off, (size_t)NGRAPHS * EMBP * 2);
    size_t o_projF = bump(off, (size_t)NGRAPHS * EMBP * 4);
    size_t o_feat  = bump(off, (size_t)FEATROWS * EMBP * 4);
    size_t o_featB = bump(off, (size_t)FEATROWS * EMBP * 2);
    size_t o_Fpk   = bump(off, (size_t)EMBP * FEATROWS * 2);
    size_t o_W1pk  = bump(off, (size_t)NLAYERS * EMBP * HIDP * 2);
    size_t o_W2pk  = bump(off, (size_t)NLAYERS * HIDP * EMBP * 2);
    size_t o_pW1pk = bump(off, (size_t)EMBP * EMBP * 2);
    size_t o_pW2pk = bump(off, (size_t)EMBP * EMBP * 2);
    size_t o_b1p   = bump(off, (size_t)NLAYERS * HIDP * 4);
    size_t o_b2p   = bump(off, (size_t)NLAYERS * EMBP * 4);
    size_t o_pb1p  = bump(off, (size_t)EMBP * 4);
    size_t o_pb2p  = bump(off, (size_t)EMBP * 4);

    float* h      = (float*)(ws + o_h);
    float* sim    = (float*)(ws + o_h);          // alias (h dead before sim)
    float* agg    = (float*)(ws + o_agg);
    unsigned short* aggB = (unsigned short*)(ws + o_aggB);
    unsigned short* midB = (unsigned short*)(ws + o_midB);
    float* stats  = (float*)(ws + o_stats);
    float* g      = (float*)(ws + o_g);
    float* cnt    = (float*)(ws + o_cnt);
    unsigned short* gB   = (unsigned short*)(ws + o_gB);
    unsigned short* p1B  = (unsigned short*)(ws + o_p1B);
    float* projF  = (float*)(ws + o_projF);
    float* feat   = (float*)(ws + o_feat);
    unsigned short* featB = (unsigned short*)(ws + o_featB);
    unsigned short* Fpk  = (unsigned short*)(ws + o_Fpk);
    unsigned short* W1pk = (unsigned short*)(ws + o_W1pk);
    unsigned short* W2pk = (unsigned short*)(ws + o_W2pk);
    unsigned short* pW1pk = (unsigned short*)(ws + o_pW1pk);
    unsigned short* pW2pk = (unsigned short*)(ws + o_pW2pk);
    float* b1p  = (float*)(ws + o_b1p);
    float* b2p  = (float*)(ws + o_b2p);
    float* pb1p = (float*)(ws + o_pb1p);
    float* pb2p = (float*)(ws + o_pb2p);

    const int T = 256;
    auto blocks = [](size_t n, int t) { return (unsigned)((n + t - 1) / t); };

    // ---- pack weights / biases (deterministic each call) ----
    for (int l = 0; l < NLAYERS; ++l) {
        pack_b<<<blocks((size_t)EMBP * HIDP, T), T, 0, stream>>>(
            W1 + (size_t)l * EMB * HID, EMB, HID, HID, 1, EMBP, HIDP,
            W1pk + (size_t)l * EMBP * HIDP);
        pack_b<<<blocks((size_t)HIDP * EMBP, T), T, 0, stream>>>(
            W2 + (size_t)l * HID * EMB, HID, EMB, EMB, 1, HIDP, EMBP,
            W2pk + (size_t)l * HIDP * EMBP);
        pad_bias<<<blocks(HIDP, T), T, 0, stream>>>(b1 + (size_t)l * HID, HID, HIDP,
                                                    b1p + (size_t)l * HIDP);
        pad_bias<<<blocks(EMBP, T), T, 0, stream>>>(b2 + (size_t)l * EMB, EMB, EMBP,
                                                    b2p + (size_t)l * EMBP);
    }
    pack_b<<<blocks((size_t)EMBP * EMBP, T), T, 0, stream>>>(pW1, EMB, EMB, EMB, 1,
                                                             EMBP, EMBP, pW1pk);
    pack_b<<<blocks((size_t)EMBP * EMBP, T), T, 0, stream>>>(pW2, EMB, EMB, EMB, 1,
                                                             EMBP, EMBP, pW2pk);
    pad_bias<<<blocks(EMBP, T), T, 0, stream>>>(pb1, EMB, EMBP, pb1p);
    pad_bias<<<blocks(EMBP, T), T, 0, stream>>>(pb2, EMB, EMBP, pb2p);

    const size_t nh = (size_t)NNODES * EMBP;
    for (int v = 0; v < 2; ++v) {
        init_h<<<blocks(nh, T), T, 0, stream>>>(x[v], atom1, atom2, h);
        for (int l = 0; l < NLAYERS; ++l) {
            const float* e1l = ee1 + (size_t)l * 6 * EMB;
            const float* e2l = ee2 + (size_t)l * 3 * EMB;
            agg_init<<<blocks(nh, T), T, 0, stream>>>(h, e1l + 4 * EMB, e2l, agg);
            scatter_edges<<<blocks((size_t)NEDGES * EMB, T), T, 0, stream>>>(
                ei[v], ea[v], h, e1l, e2l, agg);
            cvt_bf16<<<blocks(nh, T), T, 0, stream>>>(agg, aggB, nh);
            // GIN MLP: relu(agg @ W1 + b1) @ W2 + b2
            gemm_bf16_wmma<<<dim3(NNODES / 128, HIDP / 32), 128, 0, stream>>>(
                aggB, W1pk + (size_t)l * EMBP * HIDP, b1p + (size_t)l * HIDP,
                nullptr, midB, NNODES, EMBP, HIDP, HIDP, 1);
            gemm_bf16_wmma<<<dim3(NNODES / 128, EMBP / 32), 128, 0, stream>>>(
                midB, W2pk + (size_t)l * HIDP * EMBP, b2p + (size_t)l * EMBP,
                agg, nullptr, NNODES, HIDP, EMBP, EMBP, 0);
            bn_stats<<<EMB, 256, 0, stream>>>(agg, stats);
            bn_apply<<<blocks(nh, T), T, 0, stream>>>(
                agg, stats, bng + (size_t)l * EMB, bnb + (size_t)l * EMB, h,
                l < NLAYERS - 1 ? 1 : 0);
        }
        // mean pool + projector + normalize
        pool_zero<<<blocks((size_t)NGRAPHS * EMBP, T), T, 0, stream>>>(g, cnt);
        pool_scatter<<<blocks(nh, T), T, 0, stream>>>(h, batch[v], g, cnt);
        pool_div<<<blocks((size_t)NGRAPHS * EMBP, T), T, 0, stream>>>(g, cnt, gB);
        gemm_bf16_wmma<<<dim3(NGRAPHS / 128, EMBP / 32), 128, 0, stream>>>(
            gB, pW1pk, pb1p, nullptr, p1B, NGRAPHS, EMBP, EMBP, EMBP, 1);
        gemm_bf16_wmma<<<dim3(NGRAPHS / 128, EMBP / 32), 128, 0, stream>>>(
            p1B, pW2pk, pb2p, projF, nullptr, NGRAPHS, EMBP, EMBP, EMBP, 0);
        norm_rows<<<NGRAPHS, 256, 0, stream>>>(
            projF, feat + (size_t)v * NGRAPHS * EMBP,
            featB + (size_t)v * NGRAPHS * EMBP);
    }

    // sim = F @ F^T  (B = F^T: element (k,n) = feat[n*EMBP + k])
    pack_b<<<blocks((size_t)EMBP * FEATROWS, T), T, 0, stream>>>(
        feat, EMBP, FEATROWS, 1, EMBP, EMBP, FEATROWS, Fpk);
    gemm_bf16_wmma<<<dim3(FEATROWS / 128, FEATROWS / 32), 128, 0, stream>>>(
        featB, Fpk, nullptr, sim, nullptr, FEATROWS, EMBP, FEATROWS, FEATROWS, 0);

    make_logits<<<blocks((size_t)FEATROWS * (FEATROWS - 1), T), T, 0, stream>>>(sim, out);
    make_labels<<<blocks(FEATROWS, T), T, 0, stream>>>(out + (size_t)FEATROWS * (FEATROWS - 1));
}
